// Qwen3NextSparseMoeBlock_41369124995205
// MI455X (gfx1250) — matrix operand order repaired and checked by reading, not compiled, not run on hardware
//
#include <hip/hip_runtime.h>
#include <cstdint>

// ---------------- problem constants ----------------
#define T_TOK 2048   // tokens (2*1024)
#define D_DIM 2048   // hidden
#define E_NUM 64     // experts
#define K_TOP 8      // experts per token
#define F_DIM 512    // moe / shared intermediate
#define BM    128    // rows per block
#define HM    64     // rows per LDS half-tile
#define HS_P  520    // padded H row stride (elements): 520*2B = 4-bank step/row
#define HS_BYTES (HM * HS_P * 2)   // 66560 B dynamic LDS

typedef unsigned short u16;
typedef __attribute__((ext_vector_type(16))) __bf16        v16bf;
typedef __attribute__((ext_vector_type(8)))  float         v8f;
typedef __attribute__((ext_vector_type(8)))  unsigned int  v8u;

// ---------------- helpers ----------------
__device__ __forceinline__ u16 f2bf(float f) {           // fp32 -> bf16, RNE
  unsigned u = __float_as_uint(f);
  unsigned r = u + 0x7FFFu + ((u >> 16) & 1u);
  return (u16)(r >> 16);
}
__device__ __forceinline__ unsigned pack2(float lo, float hi) {   // RNE pair
  return (unsigned)f2bf(lo) | ((unsigned)f2bf(hi) << 16);
}
// round-half-up bf16 pair pack: 2 adds + 1 v_perm_b32
__device__ __forceinline__ unsigned pack2h(float lo, float hi) {
  unsigned a = __float_as_uint(lo) + 0x8000u;
  unsigned b = __float_as_uint(hi) + 0x8000u;
  return __builtin_amdgcn_perm(b, a, 0x07060302u);  // {b[31:16], a[31:16]}
}
__device__ __forceinline__ u16 f2bf_h(float f) {
  return (u16)((__float_as_uint(f) + 0x8000u) >> 16);
}
__device__ __forceinline__ v8f vzero() {
  v8f z = {0.f,0.f,0.f,0.f,0.f,0.f,0.f,0.f};
  return z;
}
// fast silu / sigmoid: v_exp_f32 + v_rcp_f32 (~1 ulp; far below bf16 noise)
__device__ __forceinline__ float silu_f(float x) {
  return x * __builtin_amdgcn_rcpf(1.f + __expf(-x));
}
__device__ __forceinline__ float sigmoid_f(float x) {
  return __builtin_amdgcn_rcpf(1.f + __expf(-x));
}
// A fragment: row fixed per lane; K = [k+off,+16), off = lane<16?0:16
__device__ __forceinline__ v8u load_a(const u16* __restrict__ row, int k, int lane) {
  int off = k + ((lane & 16) ? 16 : 0);
  uint4 a = *(const uint4*)(row + off);
  uint4 b = *(const uint4*)(row + off + 8);
  v8u r;
  r[0]=a.x; r[1]=a.y; r[2]=a.z; r[3]=a.w;
  r[4]=b.x; r[5]=b.y; r[6]=b.z; r[7]=b.w;
  return r;
}
// B fragment: 16 contiguous K values for this lane's column (bf16 direct or fp32+pack)
template<bool BF16W>
__device__ __forceinline__ v8u load_b(const void* __restrict__ W, size_t off) {
  v8u r;
  if constexpr (BF16W) {
    const u16* p = (const u16*)W + off;
    uint4 a = *(const uint4*)p;
    uint4 b = *(const uint4*)(p + 8);
    r[0]=a.x; r[1]=a.y; r[2]=a.z; r[3]=a.w;
    r[4]=b.x; r[5]=b.y; r[6]=b.z; r[7]=b.w;
  } else {
    const float* p = (const float*)W + off;
#pragma unroll
    for (int i = 0; i < 4; ++i) {
      float4 v = ((const float4*)p)[i];
      r[2*i]   = pack2h(v.x, v.y);
      r[2*i+1] = pack2h(v.z, v.w);
    }
  }
  return r;
}
__device__ __forceinline__ v8f wmma_bf16(v8u a, v8u b, v8f c) {
  return __builtin_amdgcn_wmma_f32_16x16x32_bf16(
      false, __builtin_bit_cast(v16bf, a),
      false, __builtin_bit_cast(v16bf, b),
      (short)0, c, false, false);
}

// ---------------- kernel 0: fp32 -> bf16 weight conversion (once per launch) ----
__global__ void __launch_bounds__(256)
cvt_bf16_kernel(const float* __restrict__ src, u16* __restrict__ dst, size_t n4) {
  size_t i = (size_t)blockIdx.x * blockDim.x + threadIdx.x;
  size_t stride = (size_t)gridDim.x * blockDim.x;
  for (; i < n4; i += stride) {
    float4 v = ((const float4*)src)[i];
    uint2 o;
    o.x = pack2(v.x, v.y);
    o.y = pack2(v.z, v.w);
    ((uint2*)dst)[i] = o;
  }
}

// ---------------- kernel 1: router + bf16 copy of x + shared gate ----------------
__global__ void __launch_bounds__(256)
router_kernel(const float* __restrict__ x, const float* __restrict__ rw,
              const float* __restrict__ wse,
              u16* __restrict__ xb, float* __restrict__ sgate,
              int* __restrict__ topi, float* __restrict__ topw)
{
  __shared__ float xs[D_DIM];
  __shared__ float part[256];
  __shared__ float lg[E_NUM];
  int t = blockIdx.x;
  int tid = threadIdx.x;
  const float* xr = x + (size_t)t * D_DIM;
  for (int i = tid; i < D_DIM; i += 256) {
    float v = xr[i];
    xs[i] = v;
    xb[(size_t)t * D_DIM + i] = f2bf(v);
  }
  __syncthreads();
  {
    int e = tid & 63, p = tid >> 6;
    const float* wr = rw + (size_t)e * D_DIM + p * 512;
    const float* xp = xs + p * 512;
    float s = 0.f;
    for (int d = 0; d < 512; ++d) s += xp[d] * wr[d];
    part[tid] = s;
  }
  __syncthreads();
  if (tid < E_NUM)
    lg[tid] = part[tid] + part[tid + 64] + part[tid + 128] + part[tid + 192];
  __syncthreads();
  {
    float s = 0.f;
    for (int d = tid; d < D_DIM; d += 256) s += xs[d] * wse[d];
    part[tid] = s;
  }
  __syncthreads();
  if (tid == 0) {
    float gs = 0.f;
    for (int i = 0; i < 256; ++i) gs += part[i];
    sgate[t] = sigmoid_f(gs);
    float mx = lg[0];
    for (int i = 1; i < E_NUM; ++i) mx = lg[i] > mx ? lg[i] : mx;
    float pr[E_NUM]; float Z = 0.f;
    for (int i = 0; i < E_NUM; ++i) { pr[i] = __expf(lg[i] - mx); Z += pr[i]; }
    float inv = 1.f / Z;
    for (int i = 0; i < E_NUM; ++i) pr[i] *= inv;
    int   ti[K_TOP]; float tw[K_TOP]; float wsum = 0.f;
    for (int k = 0; k < K_TOP; ++k) {
      int bi = 0; float bv = pr[0];
      for (int i = 1; i < E_NUM; ++i) if (pr[i] > bv) { bv = pr[i]; bi = i; }
      ti[k] = bi; tw[k] = bv; wsum += bv; pr[bi] = -1.f;
    }
    float winv = 1.f / wsum;
    for (int k = 0; k < K_TOP; ++k) {
      topi[t * K_TOP + k] = ti[k];
      topw[t * K_TOP + k] = tw[k] * winv;
    }
  }
}

// ---------------- kernel 2: deterministic per-expert token lists ----------------
__global__ void __launch_bounds__(64)
build_lists_kernel(const int* __restrict__ topi, const float* __restrict__ topw,
                   int* __restrict__ counts, int* __restrict__ atok,
                   float* __restrict__ aw)
{
  int e = threadIdx.x;
  int pos = 0;
  for (int t = 0; t < T_TOK; ++t) {
#pragma unroll
    for (int k = 0; k < K_TOP; ++k) {
      if (topi[t * K_TOP + k] == e) {
        atok[(size_t)e * T_TOK + pos] = t;
        aw[(size_t)e * T_TOK + pos]   = topw[t * K_TOP + k];
        ++pos;
      }
    }
  }
  counts[e] = pos;
}

// ---------------- kernel 3: shared expert (initializes out) ----------------
template<bool BF16W>
__global__ void __launch_bounds__(256)
shared_ffn_kernel(const u16* __restrict__ xb,
                  const void* __restrict__ Wsg, const void* __restrict__ Wsu,
                  const void* __restrict__ Wsd, const float* __restrict__ sgate,
                  float* __restrict__ out)
{
  extern __shared__ u16 Hs[];           // HM x F padded (HS_P)
  int row0 = blockIdx.x * BM;
  int lane = threadIdx.x & 31;
  int wave = threadIdx.x >> 5;
  int nlo  = lane & 15;
  int mhi  = (lane >> 4) << 3;
  int koff = (lane & 16) ? 16 : 0;

  for (int half = 0; half < 2; ++half) {
    int rbase = row0 + half * HM;
    const u16* ap[4];
#pragma unroll
    for (int mi = 0; mi < 4; ++mi)
      ap[mi] = xb + (size_t)(rbase + mi * 16 + nlo) * D_DIM;

    // ---- stage A: wave owns 32 F-cols per pass (2 passes) over all 4 m-tiles
    for (int fpass = 0; fpass < 2; ++fpass) {
      int fb = wave * 32 + fpass * 256;
      v8f accg[4][2], accu[4][2];
#pragma unroll
      for (int mi = 0; mi < 4; ++mi)
#pragma unroll
        for (int ni = 0; ni < 2; ++ni) { accg[mi][ni] = vzero(); accu[mi][ni] = vzero(); }
      for (int k = 0; k < D_DIM; k += 32) {
        v8u a[4];
#pragma unroll
        for (int mi = 0; mi < 4; ++mi) a[mi] = load_a(ap[mi], k, lane);
#pragma unroll
        for (int ni = 0; ni < 2; ++ni) {
          size_t boff = (size_t)(fb + 16 * ni + nlo) * D_DIM + k + koff;
          v8u bg = load_b<BF16W>(Wsg, boff);
#pragma unroll
          for (int mi = 0; mi < 4; ++mi) accg[mi][ni] = wmma_bf16(a[mi], bg, accg[mi][ni]);
          v8u bu = load_b<BF16W>(Wsu, boff);
#pragma unroll
          for (int mi = 0; mi < 4; ++mi) accu[mi][ni] = wmma_bf16(a[mi], bu, accu[mi][ni]);
        }
      }
#pragma unroll
      for (int mi = 0; mi < 4; ++mi)
#pragma unroll
        for (int ni = 0; ni < 2; ++ni)
#pragma unroll
          for (int r = 0; r < 8; ++r) {
            int M = mi * 16 + mhi + r;
            Hs[M * HS_P + fb + 16 * ni + nlo] =
                f2bf_h(silu_f(accg[mi][ni][r]) * accu[mi][ni][r]);
          }
    }
    __syncthreads();
    // ---- stage B: out = gate * (H Wd^T); wave covers 32 D-cols per pass (8 passes)
    const u16* hp[4];
#pragma unroll
    for (int mi = 0; mi < 4; ++mi) hp[mi] = &Hs[(mi * 16 + nlo) * HS_P];
    for (int npass = 0; npass < 8; ++npass) {
      int nb = wave * 32 + npass * 256;
      v8f acc[4][2];
#pragma unroll
      for (int mi = 0; mi < 4; ++mi)
#pragma unroll
        for (int ni = 0; ni < 2; ++ni) acc[mi][ni] = vzero();
      for (int k = 0; k < F_DIM; k += 32) {
        v8u a[4];
#pragma unroll
        for (int mi = 0; mi < 4; ++mi) a[mi] = load_a(hp[mi], k, lane);
#pragma unroll
        for (int ni = 0; ni < 2; ++ni) {
          v8u b = load_b<BF16W>(Wsd, (size_t)(nb + 16 * ni + nlo) * F_DIM + k + koff);
#pragma unroll
          for (int mi = 0; mi < 4; ++mi) acc[mi][ni] = wmma_bf16(a[mi], b, acc[mi][ni]);
        }
      }
#pragma unroll
      for (int mi = 0; mi < 4; ++mi)
#pragma unroll
        for (int r = 0; r < 8; ++r) {
          int tok = rbase + mi * 16 + mhi + r;
          float g = sgate[tok];
#pragma unroll
          for (int ni = 0; ni < 2; ++ni)
            out[(size_t)tok * D_DIM + nb + 16 * ni + nlo] = g * acc[mi][ni][r];
        }
    }
    __syncthreads();
  }
}

// ---------------- kernel 4: routed experts (gathered grouped GEMM) ----------------
template<bool BF16W>
__global__ void __launch_bounds__(256)
routed_ffn_kernel(const u16* __restrict__ xb,
                  const void* __restrict__ Wg, const void* __restrict__ Wu,
                  const void* __restrict__ Wd,
                  const int* __restrict__ counts, const int* __restrict__ atok,
                  const float* __restrict__ aw,
                  float* __restrict__ out)
{
  extern __shared__ u16 Hs[];
  int e    = blockIdx.x >> 4;
  int blk  = blockIdx.x & 15;
  int n_e  = counts[e];
  int row0 = blk * BM;
  if (row0 >= n_e) return;

  int lane = threadIdx.x & 31;
  int wave = threadIdx.x >> 5;
  int nlo  = lane & 15;
  int mhi  = (lane >> 4) << 3;
  int koff = (lane & 16) ? 16 : 0;

  size_t we_off = (size_t)e * F_DIM * D_DIM;   // element offsets identical for f32/bf16
  const void* wg = BF16W ? (const void*)((const u16*)Wg + we_off) : (const void*)((const float*)Wg + we_off);
  const void* wu = BF16W ? (const void*)((const u16*)Wu + we_off) : (const void*)((const float*)Wu + we_off);
  const void* wd = BF16W ? (const void*)((const u16*)Wd + we_off) : (const void*)((const float*)Wd + we_off);
  const int*   te = atok + (size_t)e * T_TOK;
  const float* we = aw   + (size_t)e * T_TOK;

  for (int half = 0; half < 2; ++half) {
    int rbase = row0 + half * HM;
    if (rbase >= n_e) break;                 // uniform across block
    const u16* ap[4];
#pragma unroll
    for (int mi = 0; mi < 4; ++mi) {
      int arow = rbase + mi * 16 + nlo;
      int arowc = arow < n_e ? arow : (n_e - 1);
      ap[mi] = xb + (size_t)te[arowc] * D_DIM;
    }
    // ---- stage A
    for (int fpass = 0; fpass < 2; ++fpass) {
      int fb = wave * 32 + fpass * 256;
      v8f accg[4][2], accu[4][2];
#pragma unroll
      for (int mi = 0; mi < 4; ++mi)
#pragma unroll
        for (int ni = 0; ni < 2; ++ni) { accg[mi][ni] = vzero(); accu[mi][ni] = vzero(); }
      for (int k = 0; k < D_DIM; k += 32) {
        v8u a[4];
#pragma unroll
        for (int mi = 0; mi < 4; ++mi) a[mi] = load_a(ap[mi], k, lane);
#pragma unroll
        for (int ni = 0; ni < 2; ++ni) {
          size_t boff = (size_t)(fb + 16 * ni + nlo) * D_DIM + k + koff;
          v8u bg = load_b<BF16W>(wg, boff);
#pragma unroll
          for (int mi = 0; mi < 4; ++mi) accg[mi][ni] = wmma_bf16(a[mi], bg, accg[mi][ni]);
          v8u bu = load_b<BF16W>(wu, boff);
#pragma unroll
          for (int mi = 0; mi < 4; ++mi) accu[mi][ni] = wmma_bf16(a[mi], bu, accu[mi][ni]);
        }
      }
#pragma unroll
      for (int mi = 0; mi < 4; ++mi)
#pragma unroll
        for (int ni = 0; ni < 2; ++ni)
#pragma unroll
          for (int r = 0; r < 8; ++r) {
            int M = mi * 16 + mhi + r;
            Hs[M * HS_P + fb + 16 * ni + nlo] =
                f2bf_h(silu_f(accg[mi][ni][r]) * accu[mi][ni][r]);
          }
    }
    __syncthreads();
    // ---- stage B: atomic scatter-add of weight-scaled down-projection
    const u16* hp[4];
#pragma unroll
    for (int mi = 0; mi < 4; ++mi) hp[mi] = &Hs[(mi * 16 + nlo) * HS_P];
    for (int npass = 0; npass < 8; ++npass) {
      int nb = wave * 32 + npass * 256;
      v8f acc[4][2];
#pragma unroll
      for (int mi = 0; mi < 4; ++mi)
#pragma unroll
        for (int ni = 0; ni < 2; ++ni) acc[mi][ni] = vzero();
      for (int k = 0; k < F_DIM; k += 32) {
        v8u a[4];
#pragma unroll
        for (int mi = 0; mi < 4; ++mi) a[mi] = load_a(hp[mi], k, lane);
#pragma unroll
        for (int ni = 0; ni < 2; ++ni) {
          v8u b = load_b<BF16W>(wd, (size_t)(nb + 16 * ni + nlo) * F_DIM + k + koff);
#pragma unroll
          for (int mi = 0; mi < 4; ++mi) acc[mi][ni] = wmma_bf16(a[mi], b, acc[mi][ni]);
        }
      }
#pragma unroll
      for (int mi = 0; mi < 4; ++mi)
#pragma unroll
        for (int r = 0; r < 8; ++r) {
          int Mg = rbase + mi * 16 + mhi + r;
          if (Mg < n_e) {
            int   tok = te[Mg];
            float w   = we[Mg];
#pragma unroll
            for (int ni = 0; ni < 2; ++ni)
              atomicAdd(&out[(size_t)tok * D_DIM + nb + 16 * ni + nlo],
                        w * acc[mi][ni][r]);
          }
        }
    }
    __syncthreads();
  }
}

// ---------------- launch ----------------
extern "C" void kernel_launch(void* const* d_in, const int* in_sizes, int n_in,
                              void* d_out, int out_size, void* d_ws, size_t ws_size,
                              hipStream_t stream) {
  (void)in_sizes; (void)n_in; (void)out_size;
  const float* x   = (const float*)d_in[0];
  const float* rw  = (const float*)d_in[1];
  const float* Wg  = (const float*)d_in[2];
  const float* Wu  = (const float*)d_in[3];
  const float* Wd  = (const float*)d_in[4];
  const float* Wsg = (const float*)d_in[5];
  const float* Wsu = (const float*)d_in[6];
  const float* Wsd = (const float*)d_in[7];
  const float* wse = (const float*)d_in[8];
  float* out = (float*)d_out;

  char* p = (char*)d_ws;
  char* pend = p + ws_size;
  u16*   xb    = (u16*)p;   p += (size_t)T_TOK * D_DIM * sizeof(u16);
  float* sgate = (float*)p; p += (size_t)T_TOK * sizeof(float);
  int*   topi  = (int*)p;   p += (size_t)T_TOK * K_TOP * sizeof(int);
  float* topw  = (float*)p; p += (size_t)T_TOK * K_TOP * sizeof(float);
  int*   counts= (int*)p;   p += (size_t)E_NUM * sizeof(int);
  p = (char*)(((uintptr_t)p + 255) & ~(uintptr_t)255);
  int*   atok  = (int*)p;   p += (size_t)E_NUM * T_TOK * sizeof(int);
  float* aw    = (float*)p; p += (size_t)E_NUM * T_TOK * sizeof(float);
  p = (char*)(((uintptr_t)p + 255) & ~(uintptr_t)255);

  const size_t NE = (size_t)E_NUM * F_DIM * D_DIM;   // per routed tensor elements
  const size_t NS = (size_t)F_DIM * D_DIM;           // per shared tensor elements
  bool preconv = ((size_t)(pend - p) >= (3 * NE + 3 * NS) * sizeof(u16));
  u16 *Wgb = nullptr, *Wub = nullptr, *Wdb = nullptr, *Wsgb = nullptr, *Wsub = nullptr, *Wsdb = nullptr;
  if (preconv) {
    Wgb  = (u16*)p; p += NE * 2;
    Wub  = (u16*)p; p += NE * 2;
    Wdb  = (u16*)p; p += NE * 2;
    Wsgb = (u16*)p; p += NS * 2;
    Wsub = (u16*)p; p += NS * 2;
    Wsdb = (u16*)p; p += NS * 2;
    cvt_bf16_kernel<<<2048, 256, 0, stream>>>(Wg,  Wgb,  NE / 4);
    cvt_bf16_kernel<<<2048, 256, 0, stream>>>(Wu,  Wub,  NE / 4);
    cvt_bf16_kernel<<<2048, 256, 0, stream>>>(Wd,  Wdb,  NE / 4);
    cvt_bf16_kernel<<<256,  256, 0, stream>>>(Wsg, Wsgb, NS / 4);
    cvt_bf16_kernel<<<256,  256, 0, stream>>>(Wsu, Wsub, NS / 4);
    cvt_bf16_kernel<<<256,  256, 0, stream>>>(Wsd, Wsdb, NS / 4);
  }

  router_kernel<<<T_TOK, 256, 0, stream>>>(x, rw, wse, xb, sgate, topi, topw);
  build_lists_kernel<<<1, 64, 0, stream>>>(topi, topw, counts, atok, aw);

  if (preconv) {
    shared_ffn_kernel<true><<<T_TOK / BM, 256, HS_BYTES, stream>>>(
        xb, Wsgb, Wsub, Wsdb, sgate, out);
    routed_ffn_kernel<true><<<E_NUM * (T_TOK / BM), 256, HS_BYTES, stream>>>(
        xb, Wgb, Wub, Wdb, counts, atok, aw, out);
  } else {
    shared_ffn_kernel<false><<<T_TOK / BM, 256, HS_BYTES, stream>>>(
        xb, Wsg, Wsu, Wsd, sgate, out);
    routed_ffn_kernel<false><<<E_NUM * (T_TOK / BM), 256, HS_BYTES, stream>>>(
        xb, Wg, Wu, Wd, counts, atok, aw, out);
  }
}
